// MoE_12790412608098
// MI455X (gfx1250) — matrix operand order repaired and checked by reading, compiled, not run on hardware
//
#include <hip/hip_runtime.h>
#include <hip/hip_bf16.h>

// MoE top-2 forward for MI455X (gfx1250), bf16 WMMA, async-LDS staging,
// double-buffered tiles. Compute-bound problem (~550 GFLOP vs ~1 GB traffic).
// ws layout (assumes ws_size >= ~437 MB):
//   [counts 16i][cursor 16i][offsets 16i] @0, data @256:
//   tk_e/tk_w/tk_slot/btok/bw (2B each, 128 KB each)
//   xb (B x D bf16)    =  32 MB
//   H  (2B x U bf16)   = 256 MB
//   O2 (2B x D f32)    = 128 MB

#define B_TOK 16384
#define DM    1024
#define NE    16
#define UH    4096
#define TM    128
#define TN    128
#define TK    32

typedef __attribute__((ext_vector_type(16))) __bf16 v16bf;
typedef __attribute__((ext_vector_type(2)))  __bf16 v2bf;
typedef __attribute__((ext_vector_type(8)))  float  v8f;
typedef __attribute__((ext_vector_type(4)))  int    v4i;

union FragBF { v16bf v; unsigned int u[8]; };

__device__ __forceinline__ unsigned short f2bf(float f) {
  unsigned int u = __float_as_uint(f);
  return (unsigned short)((u + 0x7FFFu + ((u >> 16) & 1u)) >> 16);  // RNE
}

// Pack two fp32 -> one dword of {lo=a, hi=b} bf16 (RNE).
__device__ __forceinline__ unsigned int cvt2bf(float a, float b) {
#if __has_builtin(__builtin_amdgcn_cvt_pk_bf16_f32)
  union { v2bf v; unsigned int u; } r;
  r.v = __builtin_amdgcn_cvt_pk_bf16_f32(a, b);
  return r.u;
#else
  return (unsigned int)f2bf(a) | ((unsigned int)f2bf(b) << 16);
#endif
}

// ---- CDNA5 async global->LDS copy (16B per lane), with portable fallback ----
#if __has_builtin(__builtin_amdgcn_global_load_async_to_lds_b128)
#define HAVE_ASYNC_LDS 1
typedef __attribute__((address_space(1))) v4i gv4i;   // global (__device__) AS
typedef __attribute__((address_space(3))) v4i lv4i;   // LDS (__shared__) AS
#endif

__device__ __forceinline__ void async_copy16(const void* g, void* l) {
#ifdef HAVE_ASYNC_LDS
  __builtin_amdgcn_global_load_async_to_lds_b128((gv4i*)g, (lv4i*)l, 0, 0);
#else
  *(uint4*)l = *(const uint4*)g;
#endif
}
__device__ __forceinline__ void wait_async_lds() {
#ifdef HAVE_ASYNC_LDS
#if __has_builtin(__builtin_amdgcn_s_wait_asynccnt)
  __builtin_amdgcn_s_wait_asynccnt(0);
#else
  asm volatile("s_wait_asynccnt 0x0" ::: "memory");
#endif
#endif
}

// ---------------- x fp32 -> bf16 (one pass) ----------------
__global__ __launch_bounds__(256) void moe_cvt_x(const float* __restrict__ x,
                                                 unsigned short* __restrict__ xb) {
  const size_t i4 = ((size_t)blockIdx.x * 256 + threadIdx.x) * 4;
  float4 v = *(const float4*)(x + i4);
  uint2 p; p.x = cvt2bf(v.x, v.y); p.y = cvt2bf(v.z, v.w);
  *(uint2*)(xb + i4) = p;
}

// ---------------- Router: logits, top-2, softmax(2), counts ----------------
__global__ __launch_bounds__(256) void moe_router(
    const float* __restrict__ x, const float* __restrict__ wr,
    const float* __restrict__ br, int* __restrict__ counts,
    int* __restrict__ tk_e, float* __restrict__ tk_w) {
  const int tok  = blockIdx.x * 8 + (threadIdx.x >> 5);
  const int lane = threadIdx.x & 31;
  const float* xr = x + (size_t)tok * DM;
  float acc[NE];
#pragma unroll
  for (int e = 0; e < NE; ++e) acc[e] = 0.f;
  for (int i = 0; i < DM / 32; ++i) {
    const int d = i * 32 + lane;
    const float xv = xr[d];
    const float* w = wr + (size_t)d * NE;
#pragma unroll
    for (int e = 0; e < NE; ++e) acc[e] += xv * w[e];
  }
#pragma unroll
  for (int off = 16; off > 0; off >>= 1)
#pragma unroll
    for (int e = 0; e < NE; ++e) acc[e] += __shfl_xor(acc[e], off, 32);
  if (lane == 0) {
#pragma unroll
    for (int e = 0; e < NE; ++e) acc[e] += br[e];
    float m0 = -3.4e38f; int e0 = 0;
#pragma unroll
    for (int e = 0; e < NE; ++e) if (acc[e] > m0) { m0 = acc[e]; e0 = e; }
    float m1 = -3.4e38f; int e1 = 0;
#pragma unroll
    for (int e = 0; e < NE; ++e) if (e != e0 && acc[e] > m1) { m1 = acc[e]; e1 = e; }
    const float w0 = 1.f / (1.f + __expf(m1 - m0));
    tk_e[tok * 2 + 0] = e0; tk_e[tok * 2 + 1] = e1;
    tk_w[tok * 2 + 0] = w0; tk_w[tok * 2 + 1] = 1.f - w0;
    atomicAdd(&counts[e0], 1);
    atomicAdd(&counts[e1], 1);
  }
}

__global__ void moe_zero_counts(int* counts) {
  if (threadIdx.x < NE) counts[threadIdx.x] = 0;
}

__global__ void moe_scan16(const int* __restrict__ counts,
                           int* __restrict__ offsets, int* __restrict__ cursor) {
  if (threadIdx.x == 0) {
    int s = 0;
    for (int e = 0; e < NE; ++e) { offsets[e] = s; cursor[e] = s; s += counts[e]; }
  }
}

__global__ __launch_bounds__(256) void moe_scatter(
    const int* __restrict__ tk_e, const float* __restrict__ tk_w,
    int* __restrict__ cursor, int* __restrict__ btok,
    float* __restrict__ bw, int* __restrict__ tk_slot) {
  const int t = blockIdx.x * 256 + threadIdx.x;
  if (t >= B_TOK) return;
#pragma unroll
  for (int k = 0; k < 2; ++k) {
    const int e = tk_e[t * 2 + k];
    const int p = atomicAdd(&cursor[e], 1);
    btok[p] = t;
    bw[p]   = tk_w[t * 2 + k];
    tk_slot[t * 2 + k] = p;
  }
}

// ---------------- GEMM1: H = silu(xb[bucket] @ W1[e] + b1[e]) ------------
__global__ __launch_bounds__(256) void moe_gemm1(
    const unsigned short* __restrict__ xb, const float* __restrict__ w1,
    const float* __restrict__ b1, const int* __restrict__ counts,
    const int* __restrict__ offsets, const int* __restrict__ btok,
    unsigned short* __restrict__ H) {
  const int e    = blockIdx.z;
  const int mt   = blockIdx.y;
  const int cntE = counts[e];
  if (mt * TM >= cntE) return;
  const int n0   = blockIdx.x * TN;
  const int base = offsets[e] + mt * TM;

  __shared__ __align__(16) unsigned short lA[2][TM * TK];   // [m][k] bf16
  __shared__ __align__(16) unsigned short lB[2][TN * TK];   // [n][k] bf16

  const int tid  = threadIdx.x;
  const int lane = tid & 31;
  const int wid  = tid >> 5;
  const int wm   = wid & 3;   // rows wm*32 .. +31
  const int wn   = wid >> 2;  // cols wn*64 .. +63

  const v8f z8 = {0.f,0.f,0.f,0.f,0.f,0.f,0.f,0.f};
  v8f acc[2][4];
#pragma unroll
  for (int mi = 0; mi < 2; ++mi)
#pragma unroll
    for (int ni = 0; ni < 4; ++ni) acc[mi][ni] = z8;

  // A staging: 2 threads/row; rows >= cnt clamped (never stored in epilogue)
  const int sr = tid >> 1, sp = tid & 1;
  int srow = mt * TM + sr; if (srow > cntE - 1) srow = cntE - 1;
  const unsigned short* aRow = xb + (size_t)btok[offsets[e] + srow] * DM;
  // B staging: thread -> (k pair, 8 columns)
  const int kp = tid >> 4;           // k = 2*kp, 2*kp+1
  const int bn = (tid & 15) * 8;     // n offset
  const float* wbase = w1 + (size_t)e * DM * UH;

  auto stageA = [&](int buf, int kc) {
    unsigned short* d = &lA[buf][sr * TK + sp * 16];
    async_copy16(aRow + kc + sp * 16,     d);
    async_copy16(aRow + kc + sp * 16 + 8, d + 8);
  };
  auto stageB = [&](int buf, int kc) {
    const float* r0 = wbase + (size_t)(kc + 2 * kp) * UH + (n0 + bn);
    const float* r1 = r0 + UH;
    float4 a0 = *(const float4*)(r0);
    float4 a1 = *(const float4*)(r0 + 4);
    float4 c0 = *(const float4*)(r1);
    float4 c1 = *(const float4*)(r1 + 4);
    unsigned int* d = (unsigned int*)&lB[buf][0];
    d[(bn + 0) * (TK / 2) + kp] = cvt2bf(a0.x, c0.x);
    d[(bn + 1) * (TK / 2) + kp] = cvt2bf(a0.y, c0.y);
    d[(bn + 2) * (TK / 2) + kp] = cvt2bf(a0.z, c0.z);
    d[(bn + 3) * (TK / 2) + kp] = cvt2bf(a0.w, c0.w);
    d[(bn + 4) * (TK / 2) + kp] = cvt2bf(a1.x, c1.x);
    d[(bn + 5) * (TK / 2) + kp] = cvt2bf(a1.y, c1.y);
    d[(bn + 6) * (TK / 2) + kp] = cvt2bf(a1.z, c1.z);
    d[(bn + 7) * (TK / 2) + kp] = cvt2bf(a1.w, c1.w);
  };

  const int rsel = lane & 15, hi = lane >> 4;
  stageA(0, 0); stageB(0, 0);

  const int KITER = DM / TK;
  for (int it = 0; it < KITER; ++it) {
    wait_async_lds();
    __syncthreads();
    const int cur = it & 1;
    if (it + 1 < KITER) { stageA(cur ^ 1, (it + 1) * TK); stageB(cur ^ 1, (it + 1) * TK); }

    const unsigned int* lA32 = (const unsigned int*)&lA[cur][0];
    const unsigned int* lB32 = (const unsigned int*)&lB[cur][0];
    FragBF aF[2], bF[4];
#pragma unroll
    for (int mi = 0; mi < 2; ++mi) {
      const int r = wm * 32 + mi * 16 + rsel;
#pragma unroll
      for (int i = 0; i < 4; ++i) {
        aF[mi].u[i]     = lA32[r * 16 + hi * 4 + i];       // K = khalf+2i
        aF[mi].u[4 + i] = lA32[r * 16 + 8 + hi * 4 + i];   // K = 16+khalf+2i
      }
    }
#pragma unroll
    for (int ni = 0; ni < 4; ++ni) {
      const int c = wn * 64 + ni * 16 + rsel;
#pragma unroll
      for (int i = 0; i < 8; ++i) bF[ni].u[i] = lB32[c * 16 + (i >> 2) * 8 + hi * 4 + (i & 3)];
    }
#pragma unroll
    for (int mi = 0; mi < 2; ++mi)
#pragma unroll
      for (int ni = 0; ni < 4; ++ni)
        acc[mi][ni] = __builtin_amdgcn_wmma_f32_16x16x32_bf16(
            false, aF[mi].v, false, bF[ni].v, (short)0, acc[mi][ni], false, false);
  }

  const int mb = hi * 8;
#pragma unroll
  for (int mi = 0; mi < 2; ++mi)
#pragma unroll
    for (int ni = 0; ni < 4; ++ni) {
      const int col  = n0 + wn * 64 + ni * 16 + rsel;
      const float bias = b1[e * UH + col];
#pragma unroll
      for (int i = 0; i < 8; ++i) {
        const int mrow = wm * 32 + mi * 16 + mb + i;
        if (mt * TM + mrow < cntE) {
          const float a = acc[mi][ni][i] + bias;
          const float s = a / (1.f + __expf(-a));          // SiLU
          H[(size_t)(base + mrow) * UH + col] = f2bf(s);
        }
      }
    }
}

// ---------------- GEMM2: O2 = we * (H @ W2[e] + b2[e]) per slot ----------
__global__ __launch_bounds__(256) void moe_gemm2(
    const unsigned short* __restrict__ H, const float* __restrict__ w2,
    const float* __restrict__ b2, const int* __restrict__ counts,
    const int* __restrict__ offsets, const float* __restrict__ bw,
    float* __restrict__ O2) {
  const int e    = blockIdx.z;
  const int mt   = blockIdx.y;
  const int cntE = counts[e];
  if (mt * TM >= cntE) return;
  const int n0   = blockIdx.x * TN;
  const int base = offsets[e] + mt * TM;

  __shared__ __align__(16) unsigned short lA[2][TM * TK];
  __shared__ __align__(16) unsigned short lB[2][TN * TK];
  __shared__ float sWe[TM];

  const int tid  = threadIdx.x;
  const int lane = tid & 31;
  const int wid  = tid >> 5;
  const int wm   = wid & 3;
  const int wn   = wid >> 2;

  if (tid < TM) sWe[tid] = (mt * TM + tid < cntE) ? bw[base + tid] : 0.f;

  const v8f z8 = {0.f,0.f,0.f,0.f,0.f,0.f,0.f,0.f};
  v8f acc[2][4];
#pragma unroll
  for (int mi = 0; mi < 2; ++mi)
#pragma unroll
    for (int ni = 0; ni < 4; ++ni) acc[mi][ni] = z8;

  const int sr = tid >> 1, sp = tid & 1;
  int srow = mt * TM + sr; if (srow > cntE - 1) srow = cntE - 1;
  const unsigned short* aRow = H + (size_t)(offsets[e] + srow) * UH;
  const int kp = tid >> 4;
  const int bn = (tid & 15) * 8;
  const float* wbase = w2 + (size_t)e * UH * DM;

  auto stageA = [&](int buf, int kc) {
    unsigned short* d = &lA[buf][sr * TK + sp * 16];
    async_copy16(aRow + kc + sp * 16,     d);
    async_copy16(aRow + kc + sp * 16 + 8, d + 8);
  };
  auto stageB = [&](int buf, int kc) {
    const float* r0 = wbase + (size_t)(kc + 2 * kp) * DM + (n0 + bn);
    const float* r1 = r0 + DM;
    float4 a0 = *(const float4*)(r0);
    float4 a1 = *(const float4*)(r0 + 4);
    float4 c0 = *(const float4*)(r1);
    float4 c1 = *(const float4*)(r1 + 4);
    unsigned int* d = (unsigned int*)&lB[buf][0];
    d[(bn + 0) * (TK / 2) + kp] = cvt2bf(a0.x, c0.x);
    d[(bn + 1) * (TK / 2) + kp] = cvt2bf(a0.y, c0.y);
    d[(bn + 2) * (TK / 2) + kp] = cvt2bf(a0.z, c0.z);
    d[(bn + 3) * (TK / 2) + kp] = cvt2bf(a0.w, c0.w);
    d[(bn + 4) * (TK / 2) + kp] = cvt2bf(a1.x, c1.x);
    d[(bn + 5) * (TK / 2) + kp] = cvt2bf(a1.y, c1.y);
    d[(bn + 6) * (TK / 2) + kp] = cvt2bf(a1.z, c1.z);
    d[(bn + 7) * (TK / 2) + kp] = cvt2bf(a1.w, c1.w);
  };

  const int rsel = lane & 15, hi = lane >> 4;
  stageA(0, 0); stageB(0, 0);

  const int KITER = UH / TK;
  for (int it = 0; it < KITER; ++it) {
    wait_async_lds();
    __syncthreads();
    const int cur = it & 1;
    if (it + 1 < KITER) { stageA(cur ^ 1, (it + 1) * TK); stageB(cur ^ 1, (it + 1) * TK); }

    const unsigned int* lA32 = (const unsigned int*)&lA[cur][0];
    const unsigned int* lB32 = (const unsigned int*)&lB[cur][0];
    FragBF aF[2], bF[4];
#pragma unroll
    for (int mi = 0; mi < 2; ++mi) {
      const int r = wm * 32 + mi * 16 + rsel;
#pragma unroll
      for (int i = 0; i < 4; ++i) {
        aF[mi].u[i]     = lA32[r * 16 + hi * 4 + i];
        aF[mi].u[4 + i] = lA32[r * 16 + 8 + hi * 4 + i];
      }
    }
#pragma unroll
    for (int ni = 0; ni < 4; ++ni) {
      const int c = wn * 64 + ni * 16 + rsel;
#pragma unroll
      for (int i = 0; i < 8; ++i) bF[ni].u[i] = lB32[c * 16 + (i >> 2) * 8 + hi * 4 + (i & 3)];
    }
#pragma unroll
    for (int mi = 0; mi < 2; ++mi)
#pragma unroll
      for (int ni = 0; ni < 4; ++ni)
        acc[mi][ni] = __builtin_amdgcn_wmma_f32_16x16x32_bf16(
            false, aF[mi].v, false, bF[ni].v, (short)0, acc[mi][ni], false, false);
  }

  const int mb = hi * 8;
#pragma unroll
  for (int mi = 0; mi < 2; ++mi)
#pragma unroll
    for (int ni = 0; ni < 4; ++ni) {
      const int col  = n0 + wn * 64 + ni * 16 + rsel;
      const float bias = b2[e * DM + col];
#pragma unroll
      for (int i = 0; i < 8; ++i) {
        const int mrow = wm * 32 + mi * 16 + mb + i;
        if (mt * TM + mrow < cntE)
          O2[(size_t)(base + mrow) * DM + col] = sWe[mrow] * (acc[mi][ni][i] + bias);
      }
    }
}

// ---------------- Combine: out = x + O2[slot0] + O2[slot1] ----------------
__global__ __launch_bounds__(256) void moe_combine(
    const float* __restrict__ x, const float* __restrict__ O2,
    const int* __restrict__ tk_slot, float* __restrict__ out) {
  const size_t idx = (size_t)blockIdx.x * 256 + threadIdx.x;
  const int t = (int)(idx >> 10);       // DM == 1024
  const int d = (int)(idx & 1023);
  const int s0 = tk_slot[2 * t + 0];
  const int s1 = tk_slot[2 * t + 1];
  out[idx] = x[idx] + O2[(size_t)s0 * DM + d] + O2[(size_t)s1 * DM + d];
}

extern "C" void kernel_launch(void* const* d_in, const int* in_sizes, int n_in,
                              void* d_out, int out_size, void* d_ws, size_t ws_size,
                              hipStream_t stream) {
  const float* x  = (const float*)d_in[0];
  const float* wr = (const float*)d_in[1];
  const float* br = (const float*)d_in[2];
  const float* w1 = (const float*)d_in[3];
  const float* b1 = (const float*)d_in[4];
  const float* w2 = (const float*)d_in[5];
  const float* b2 = (const float*)d_in[6];
  float* out = (float*)d_out;

  char* ws = (char*)d_ws;
  int*   counts  = (int*)ws;         // 16
  int*   cursor  = counts + 16;      // 16
  int*   offsets = cursor + 16;      // 16
  size_t o = 256;
  int*   tk_e    = (int*)(ws + o);   o += (size_t)B_TOK * 2 * 4;
  float* tk_w    = (float*)(ws + o); o += (size_t)B_TOK * 2 * 4;
  int*   tk_slot = (int*)(ws + o);   o += (size_t)B_TOK * 2 * 4;
  int*   btok    = (int*)(ws + o);   o += (size_t)B_TOK * 2 * 4;
  float* bw      = (float*)(ws + o); o += (size_t)B_TOK * 2 * 4;
  unsigned short* xb = (unsigned short*)(ws + o);
  o += (size_t)B_TOK * DM * 2;                           //  32 MB
  unsigned short* H = (unsigned short*)(ws + o);
  o += (size_t)B_TOK * 2 * UH * 2;                       // 256 MB
  float* O2 = (float*)(ws + o);                          // 128 MB

  moe_cvt_x<<<(B_TOK * DM / 4) / 256, 256, 0, stream>>>(x, xb);
  moe_zero_counts<<<1, 32, 0, stream>>>(counts);
  moe_router<<<B_TOK / 8, 256, 0, stream>>>(x, wr, br, counts, tk_e, tk_w);
  moe_scan16<<<1, 32, 0, stream>>>(counts, offsets, cursor);
  moe_scatter<<<B_TOK / 256, 256, 0, stream>>>(tk_e, tk_w, cursor, btok, bw, tk_slot);
  moe_gemm1<<<dim3(UH / TN, B_TOK / TM, NE), 256, 0, stream>>>(
      xb, w1, b1, counts, offsets, btok, H);
  moe_gemm2<<<dim3(DM / TN, B_TOK / TM, NE), 256, 0, stream>>>(
      H, w2, b2, counts, offsets, bw, O2);
  moe_combine<<<(B_TOK * DM) / 256, 256, 0, stream>>>(x, O2, tk_slot, out);
}
